// EmbeddingBlock_77146202571329
// MI455X (gfx1250) — matrix operand order repaired and checked by reading, not compiled, not run on hardware
//
#include <hip/hip_runtime.h>

// ---------------------------------------------------------------------------
// DimeNet-style EmbeddingBlock on MI455X (gfx1250).
// out[e] = swish(concat(emb[Z[i(e)]], emb[Z[j(e)]], swish(rbf[e]@W_rbf+b_rbf)) @ W + b)
//
// Roofline: 98 GFLOP main GEMM vs ~550MB HBM traffic (512MB = f32 output).
// Memory floor ~24us @ 23.3 TB/s; f16 WMMA (f32 accum) keeps compute below
// that, scalar f32 FMA would not. So: v_wmma_f32_16x16x32_f16 for h@W,
// VALU for the 6-wide rbf layer + swish epilogues.
//
// v2 changes vs v1 (from asm inspection):
//  - Software-pipelined B-fragment LDS loads so v_wmma no longer sits behind
//    s_wait_dscnt 0x0 every step.
//  - Bias preloaded outside the N loop (was a global_load_b32 + waitcnt in
//    the loop header).
//  - TM 128->64: LDS 192KB->144KB, so 2 workgroups (16 waves) fit per WGP's
//    320KB for cross-block phase overlap.
//  - Staging now packs 8 halves per ds_store_b128 (swizzles keep 8
//    consecutive K contiguous per lane) and uses float4 global loads.
// ---------------------------------------------------------------------------

#define F    128          // features
#define R    6            // radial
#define K3   384          // 3*F  (concat K dimension)
#define TM   64           // edges per block
#define MT   4            // M tiles of 16 (TM/16)
#define KT   12           // K tiles of 32 (K3/32)
#define NT   8            // N tiles of 16 (F/16)

typedef __attribute__((ext_vector_type(16))) _Float16 v16h;
typedef __attribute__((ext_vector_type(8)))  _Float16 v8h;
typedef __attribute__((ext_vector_type(8)))  float    v8f;

__device__ __forceinline__ float swishf(float x) {
  return x / (1.0f + __expf(-x));
}

// Fragment-major LDS layout: frag = [32 lanes][16 halves] = 1 KiB; each WMMA
// operand is one contiguous 32B LDS load per lane.
//   A frags: tile (mt,kt) at index mt*KT+kt   -> MT*KT KiB = 48 KiB
//   B frags: tile (kt,nt) at index kt*NT+nt   -> KT*NT KiB = 96 KiB
__global__ __launch_bounds__(256)
void embed_block_wmma(const int* __restrict__ Zt,  const float* __restrict__ rbf,
                      const int* __restrict__ nbi, const int* __restrict__ nbj,
                      const float* __restrict__ emb, const float* __restrict__ Wr,
                      const float* __restrict__ br,  const float* __restrict__ W,
                      const float* __restrict__ bv,  float* __restrict__ out, int E)
{
  __shared__ _Float16 Asw[MT * KT * 32 * 16];   // 48 KiB
  __shared__ _Float16 Bsw[KT * NT * 32 * 16];   // 96 KiB

  const int  t     = threadIdx.x;
  const long tile0 = (long)blockIdx.x * TM;

  // ---- Phase 0: W (384x128 f32, row-major) -> f16 B fragments.
  // 16-bit B 32x16 (this kernel's k order): lanes 0-15 hold K=0..15 of col N=lane, lanes 16-31
  // hold K=16..31. 8 consecutive K land in one lane's contiguous halves, so
  // pack 8 and do one ds_store_b128. Groups: 48 k-chunks x 128 cols.
  for (int g = t; g < (K3 / 8) * F; g += 256) {
    const int kb = g >> 7;          // k-chunk (0..47), k0 = kb*8
    const int c  = g & 127;         // column (coalesced across threads)
    const int k0 = kb << 3;
    const int kt = k0 >> 5, kk0 = k0 & 31;
    const int nt = c >> 4,  n   = c & 15;
    const int lane = n + ((kk0 >> 4) << 4);
    const int h0   = kk0 & 15;      // 0 or 8
    v8h pk;
#pragma unroll
    for (int i = 0; i < 8; ++i) pk[i] = (_Float16)W[(k0 + i) * F + c];
    *(v8h*)&Bsw[((kt * NT + nt) * 32 + lane) * 16 + h0] = pk;
  }

  // ---- Phase 1: A rows = [emb[z_i] | emb[z_j] | swish(rbf@Wr+br)] into the
  // 16-bit A 16x32 fragment swizzle: lane = row + (kk>>4)*16,
  // h = kk&15. 8 consecutive k -> contiguous halves.
  {
    const int r  = t >> 2;                // edge row in tile (0..63)
    const int p  = t & 3;                 // covers k in [p*96, p*96+96)
    long e = tile0 + r; if (e >= E) e = E - 1;   // clamp for partial last tile
    const int z1 = Zt[nbi[e]], z2 = Zt[nbj[e]];
    const float* e1 = emb + (long)z1 * F;
    const float* e2 = emb + (long)z2 * F;
    float rv[R];
#pragma unroll
    for (int j = 0; j < R; ++j) rv[j] = rbf[e * R + j];
    const int mt = r >> 4, row = r & 15;
#pragma unroll 1
    for (int kb = 0; kb < 12; ++kb) {           // 12 chunks of 8 k-values
      const int k = p * 96 + kb * 8;
      v8h pk;
      if (k < 2 * F) {
        const float* src = (k < F) ? (e1 + k) : (e2 + (k - F));
        const float4 u0 = *(const float4*)(src);
        const float4 u1 = *(const float4*)(src + 4);
        pk[0] = (_Float16)u0.x; pk[1] = (_Float16)u0.y;
        pk[2] = (_Float16)u0.z; pk[3] = (_Float16)u0.w;
        pk[4] = (_Float16)u1.x; pk[5] = (_Float16)u1.y;
        pk[6] = (_Float16)u1.z; pk[7] = (_Float16)u1.w;
      } else {
#pragma unroll
        for (int i = 0; i < 8; ++i) {
          const int f = k + i - 2 * F;
          float s = br[f];
#pragma unroll
          for (int j = 0; j < R; ++j) s += rv[j] * Wr[j * F + f];
          pk[i] = (_Float16)swishf(s);
        }
      }
      const int kt   = k >> 5, kk = k & 31;
      const int lane = row + ((kk >> 4) << 4);
      const int h0   = kk & 15;   // 0 or 8 (k is 8-aligned)
      *(v8h*)&Asw[((mt * KT + kt) * 32 + lane) * 16 + h0] = pk;
    }
  }
  __syncthreads();

  // ---- Phase 2: wave (mt, nhalf) computes a 16-edge x 64-feature tile.
  const int wave  = t >> 5;
  const int lane  = t & 31;
  const int mt    = wave >> 1;        // M tile (0..3)
  const int nhalf = wave & 1;         // N tiles [nhalf*4, nhalf*4+4)
  const v16h* Af = (const v16h*)Asw;  // one v16h == one lane's 32B frag slice
  const v16h* Bf = (const v16h*)Bsw;

  // All 12 A fragments live in VGPRs across the N loop.
  v16h afr[KT];
#pragma unroll
  for (int kt = 0; kt < KT; ++kt) afr[kt] = Af[(mt * KT + kt) * 32 + lane];

  // Preload biases (epilogue) so no global load sits in the N loop header.
  float bbv[4];
#pragma unroll
  for (int i = 0; i < 4; ++i) bbv[i] = bv[(nhalf * 4 + i) * 16 + (lane & 15)];

#pragma unroll 1
  for (int i = 0; i < 4; ++i) {
    const int nt = nhalf * 4 + i;
    v8f acc = {};
    // Software-pipeline B: issue load for kt+1 before the wmma for kt.
    v16h bcur = Bf[(0 * NT + nt) * 32 + lane];
#pragma unroll
    for (int kt = 0; kt < KT; ++kt) {
      v16h bnext = bcur;
      if (kt + 1 < KT) bnext = Bf[((kt + 1) * NT + nt) * 32 + lane];
      // (neg_a, A, neg_b, B, c_mod, C, reuse_a, reuse_b)
      acc = __builtin_amdgcn_wmma_f32_16x16x32_f16(
          false, afr[kt], false, bcur, (short)0, acc, false, false);
      bcur = bnext;
    }
    // Epilogue: C layout (16x16 f32): VGPR r -> M=r (lanes 0-15) / M=r+8
    // (lanes 16-31), N = lane%16.
    const int   feat = nt * 16 + (lane & 15);
    const float bb   = bbv[i];
    const int   mofs = (lane >> 4) << 3;
#pragma unroll
    for (int ri = 0; ri < 8; ++ri) {
      const long e = tile0 + (long)mt * 16 + ri + mofs;
      if (e < E) {
        const float x = acc[ri] + bb;
        out[e * F + feat] = swishf(x);
      }
    }
  }
}

extern "C" void kernel_launch(void* const* d_in, const int* in_sizes, int n_in,
                              void* d_out, int out_size, void* d_ws, size_t ws_size,
                              hipStream_t stream) {
  const int*   Zt  = (const int*)  d_in[0];
  const float* rbf = (const float*)d_in[1];
  const int*   nbi = (const int*)  d_in[2];
  const int*   nbj = (const int*)  d_in[3];
  const float* emb = (const float*)d_in[4];
  const float* Wr  = (const float*)d_in[5];
  const float* br  = (const float*)d_in[6];
  const float* W   = (const float*)d_in[7];
  const float* bv  = (const float*)d_in[8];
  float*       out = (float*)d_out;
  const int E = in_sizes[2];                 // idnb_i has one entry per edge
  const int blocks = (E + TM - 1) / TM;
  embed_block_wmma<<<blocks, 256, 0, stream>>>(Zt, rbf, nbi, nbj, emb, Wr, br, W,
                                               bv, out, E);
  (void)d_ws; (void)ws_size; (void)out_size; (void)n_in;
}
